// DiscoMLPAgent_3779571221144
// MI455X (gfx1250) — matrix operand order repaired and checked by reading, not compiled, not run on hardware
//
#include <hip/hip_runtime.h>
#include <hip/hip_bf16.h>

// ---------------------------------------------------------------------------
// DiscoMLPAgent forward on MI455X (gfx1250).
// bf16 WMMA (v_wmma_f32_16x16x32_bf16, f32 accumulate) for every GEMM;
// A tiles staged via global_load_async_to_lds_b128 (ASYNCcnt) with double-
// buffered LDS; B (weights) pre-packed to WMMA pair layout and streamed
// straight from the 192MB L2. Pointwise LSTM math stays f32 on the VALU.
// Workload is output-bandwidth bound (~658MB f32 out, ~90 GFLOP).
// ---------------------------------------------------------------------------

typedef __attribute__((ext_vector_type(16))) __bf16 v16bf;
typedef __attribute__((ext_vector_type(8)))  float  v8f;

union FragBF { unsigned int u[8]; v16bf v; };

__device__ __forceinline__ unsigned short f2bf(float f) {
  unsigned u = __float_as_uint(f);
  u += 0x7FFFu + ((u >> 16) & 1u);   // round-to-nearest-even
  return (unsigned short)(u >> 16);
}

// Async global->LDS copy, 128b per lane. LDS offset = low 32 bits of generic
// shared pointer (ISA: LDS aperture address = addr[31:0]). Tracked by ASYNCcnt.
__device__ __forceinline__ void async_copy_b128(const unsigned short* g,
                                                const unsigned short* l) {
  unsigned loff = (unsigned)(unsigned long long)l;
  asm volatile("global_load_async_to_lds_b128 %0, %1, off"
               :: "v"(loff), "v"(g) : "memory");
}
__device__ __forceinline__ void wait_async0() {
  asm volatile("s_wait_asynccnt 0x0" ::: "memory");
}

// ---------------- f32 -> bf16 row-major (activations / obs) ----------------
__global__ void convert_bf16_kernel(const float* __restrict__ src,
                                    unsigned short* __restrict__ dst, int total) {
  int i = blockIdx.x * blockDim.x + threadIdx.x;
  if (i >= total) return;
  dst[i] = f2bf(src[i]);
}

// ------- f32 W[K,N] -> bf16 pair-interleaved [K/2][Npad][2], zero padded ----
// Reader: lane pulls u32 at element ((kp*Npad + n)*2) = {k=2kp, k=2kp+1}.
__global__ void convert_pack_w_kernel(const float* __restrict__ src,
                                      unsigned short* __restrict__ dst,
                                      int K, int N, int Npad) {
  int o = blockIdx.x * blockDim.x + threadIdx.x;
  int total = K * Npad;
  if (o >= total) return;
  int kl = o & 1;
  int p  = o >> 1;
  int n  = p % Npad;
  int kp = p / Npad;
  int k  = kp * 2 + kl;
  float v = (n < N) ? src[(size_t)k * N + n] : 0.0f;
  dst[o] = f2bf(v);
}

// ---------------- generic bf16 WMMA GEMM -----------------------------------
//   C = act(A[M,K](bf16 row-major) @ W2(pair-packed [K/2][Npad][2]) + bias)
// mode 0: bf16 row-major out (ldo=Npad, Nlog==Npad)
// mode 1: f32 row-major out (clamp col<Nlog)
// mode 2: f32 "head scatter": row r=a*8192+n -> out[(n*16+a)*ldo + col]
__global__ __launch_bounds__(128)
void gemm_wmma_bf16(const unsigned short* __restrict__ A, int lda,
                    const unsigned short* __restrict__ W2, int ldw,
                    const float* __restrict__ bias,
                    float* __restrict__ outF,
                    unsigned short* __restrict__ outB,
                    int K, int Nlog, int ldo, int relu, int mode) {
  __shared__ unsigned short lA[2][64][40];   // 80B rows -> 16B-aligned chunks

  const int t      = threadIdx.x;
  const int wave   = t >> 5;
  const int lane   = t & 31;
  const int laneLo = lane & 15;
  const int hi     = lane >> 4;
  const int m0     = blockIdx.y * 64;
  const int n0     = blockIdx.x * 64;
  const int wm     = (wave >> 1) * 32;
  const int wn     = (wave & 1) * 32;

  // A staging: 64x32 bf16 tile = 4KB; 2 x 16B chunks per thread, async.
  const int c0 = t * 2, c1 = t * 2 + 1;
  const int ar0 = c0 >> 2, ak0 = (c0 & 3) << 3;
  const int ar1 = c1 >> 2, ak1 = (c1 & 3) << 3;
  const unsigned short* gA0 = A + (size_t)(m0 + ar0) * lda + ak0;
  const unsigned short* gA1 = A + (size_t)(m0 + ar1) * lda + ak1;

  v8f acc[2][2];
#pragma unroll
  for (int i = 0; i < 2; ++i)
#pragma unroll
    for (int j = 0; j < 2; ++j)
#pragma unroll
      for (int v = 0; v < 8; ++v) acc[i][j][v] = 0.0f;

  async_copy_b128(gA0, &lA[0][ar0][ak0]);
  async_copy_b128(gA1, &lA[0][ar1][ak1]);

  const int ncol0 = n0 + wn + laneLo;   // B column for this lane, frag j adds 16
  int cur = 0;
  for (int kt = 0; kt < K; kt += 32, cur ^= 1) {
    wait_async0();           // my async copies for lA[cur] landed
    __syncthreads();         // everyone's landed; prior readers of lA[cur^1] done
    if (kt + 32 < K) {       // overlap next tile's DMA with this tile's WMMAs
      async_copy_b128(gA0 + kt + 32, &lA[cur ^ 1][ar0][ak0]);
      async_copy_b128(gA1 + kt + 32, &lA[cur ^ 1][ar1][ak1]);
    }

    // A fragments from LDS (contiguous k = 16*hi + s order, same as B below)
    FragBF aF[2], bF[2];
#pragma unroll
    for (int i = 0; i < 2; ++i) {
      int m = wm + i * 16 + laneLo;
#pragma unroll
      for (int r = 0; r < 8; ++r) {
        int k = 2 * r + (hi ? 16 : 0);
        aF[i].u[r] = *(const unsigned int*)&lA[cur][m][k];
      }
    }
    // B fragments straight from pair-packed weights in L2 (coalesced b32)
#pragma unroll
    for (int j = 0; j < 2; ++j) {
      int ncol = ncol0 + j * 16;
#pragma unroll
      for (int r = 0; r < 8; ++r) {
        size_t kp = (size_t)((kt >> 1) + (hi ? 8 : 0) + r);
        bF[j].u[r] = *(const unsigned int*)(W2 + (kp * (size_t)ldw + ncol) * 2);
      }
    }

#pragma unroll
    for (int i = 0; i < 2; ++i)
#pragma unroll
      for (int j = 0; j < 2; ++j)
        acc[i][j] = __builtin_amdgcn_wmma_f32_16x16x32_bf16(
            false, aF[i].v, false, bF[j].v, (short)0, acc[i][j], false, false);
  }

  // ---- epilogue: bias + activation + store ----
#pragma unroll
  for (int i = 0; i < 2; ++i) {
#pragma unroll
    for (int j = 0; j < 2; ++j) {
      int col  = n0 + wn + j * 16 + laneLo;
      bool cok = col < Nlog;
      float bv = (bias != nullptr && cok) ? bias[col] : 0.0f;
#pragma unroll
      for (int v = 0; v < 8; ++v) {
        int row = m0 + wm + i * 16 + v + (hi ? 8 : 0);
        float x = acc[i][j][v] + bv;
        if (relu) x = fmaxf(x, 0.0f);
        if (mode == 0) {
          outB[(size_t)row * ldo + col] = f2bf(x);
        } else if (cok) {
          size_t orow = (mode == 1)
                            ? (size_t)row
                            : ((size_t)(row & 8191) * 16 + (size_t)(row >> 13));
          outF[orow * (size_t)ldo + col] = x;
        }
      }
    }
  }
}

// ---------------- h0 = tanh(c0), bf16 copy for the recurrence --------------
__global__ void tanh_c0_kernel(const float* __restrict__ c,
                               unsigned short* __restrict__ h0, int n) {
  int i = blockIdx.x * blockDim.x + threadIdx.x;
  if (i >= n) return;
  h0[i] = f2bf(tanhf(c[i]));
}

// ---------------- LSTM pointwise gate update -------------------------------
__global__ void lstm_pw_kernel(const float* __restrict__ G,      // [8192,512]
                               const float* __restrict__ w_ih,   // [16,512]
                               const float* __restrict__ b_ih,   // [512]
                               const float* __restrict__ b_hh,   // [512]
                               int a,
                               float* __restrict__ c,            // [8192,128] rw
                               unsigned short* __restrict__ hout)// h_all slot
{
  int i = blockIdx.x * blockDim.x + threadIdx.x;
  if (i >= 8192 * 128) return;
  int n = i >> 7;
  int h = i & 127;
  const float* g  = G + (size_t)n * 512;
  const float* wr = w_ih + (size_t)a * 512;
  float gi = g[h]       + wr[h]       + b_ih[h]       + b_hh[h];
  float gf = g[128 + h] + wr[128 + h] + b_ih[128 + h] + b_hh[128 + h];
  float gg = g[256 + h] + wr[256 + h] + b_ih[256 + h] + b_hh[256 + h];
  float go = g[384 + h] + wr[384 + h] + b_ih[384 + h] + b_hh[384 + h];
  float si = 1.0f / (1.0f + __expf(-gi));
  float sf = 1.0f / (1.0f + __expf(-gf));
  float so = 1.0f / (1.0f + __expf(-go));
  float cn = sf * c[i] + si * tanhf(gg);
  float hn = so * tanhf(cn);
  c[i]    = cn;
  hout[i] = f2bf(hn);
}

// ---------------------------------------------------------------------------
extern "C" void kernel_launch(void* const* d_in, const int* in_sizes, int n_in,
                              void* d_out, int out_size, void* d_ws, size_t ws_size,
                              hipStream_t stream) {
  (void)in_sizes; (void)n_in; (void)out_size; (void)ws_size;
  constexpr int N8 = 8192;        // T*B
  constexpr int NH = 16 * N8;     // A*T*B rows for head MLPs

  const float* obs   = (const float*)d_in[0];
  const float* bbw1  = (const float*)d_in[1];  const float* bbb1 = (const float*)d_in[2];
  const float* bbw2  = (const float*)d_in[3];  const float* bbb2 = (const float*)d_in[4];
  const float* polw  = (const float*)d_in[5];  const float* polb = (const float*)d_in[6];
  const float* yw    = (const float*)d_in[7];  const float* yb   = (const float*)d_in[8];
  const float* ciw   = (const float*)d_in[9];  const float* cib  = (const float*)d_in[10];
  const float* wih   = (const float*)d_in[11];
  const float* whh   = (const float*)d_in[12];
  const float* bih   = (const float*)d_in[13]; const float* bhh  = (const float*)d_in[14];
  const float* zw1   = (const float*)d_in[15]; const float* zb1  = (const float*)d_in[16];
  const float* zw2   = (const float*)d_in[17]; const float* zb2  = (const float*)d_in[18];
  const float* qw1   = (const float*)d_in[19]; const float* qb1  = (const float*)d_in[20];
  const float* qw2   = (const float*)d_in[21]; const float* qb2  = (const float*)d_in[22];
  const float* aw1   = (const float*)d_in[23]; const float* ab1  = (const float*)d_in[24];
  const float* aw2   = (const float*)d_in[25]; const float* ab2  = (const float*)d_in[26];
  float* out = (float*)d_out;

  // ---- workspace sub-allocation (256B aligned) ----
  char* ws = (char*)d_ws;
  size_t off = 0;
  auto alloc = [&](size_t bytes) -> char* {
    char* p = ws + off;
    off = (off + bytes + 255) & ~(size_t)255;
    return p;
  };
  unsigned short* obs_bf = (unsigned short*)alloc((size_t)N8 * 1024 * 2);
  unsigned short* wb_bb1 = (unsigned short*)alloc(1024 * 512 * 2);
  unsigned short* wb_bb2 = (unsigned short*)alloc(512 * 512 * 2);
  unsigned short* wb_pol = (unsigned short*)alloc(512 * 64 * 2);
  unsigned short* wb_y   = (unsigned short*)alloc(512 * 640 * 2);
  unsigned short* wb_ci  = (unsigned short*)alloc(512 * 128 * 2);
  unsigned short* wb_hh  = (unsigned short*)alloc(128 * 512 * 2);
  unsigned short* wb_z1  = (unsigned short*)alloc(128 * 128 * 2);
  unsigned short* wb_z2  = (unsigned short*)alloc(128 * 640 * 2);
  unsigned short* wb_q1  = (unsigned short*)alloc(128 * 128 * 2);
  unsigned short* wb_q2  = (unsigned short*)alloc(128 * 640 * 2);
  unsigned short* wb_a1  = (unsigned short*)alloc(128 * 128 * 2);
  unsigned short* wb_a2  = (unsigned short*)alloc(128 * 64 * 2);
  unsigned short* h1_bf  = (unsigned short*)alloc((size_t)N8 * 512 * 2);
  unsigned short* emb_bf = (unsigned short*)alloc((size_t)N8 * 512 * 2);
  float*          c_f32  = (float*)alloc((size_t)N8 * 128 * 4);
  unsigned short* h0_bf  = (unsigned short*)alloc((size_t)N8 * 128 * 2);
  float*          G_f32  = (float*)alloc((size_t)N8 * 512 * 4);
  unsigned short* h_all  = (unsigned short*)alloc((size_t)NH * 128 * 2);
  unsigned short* hid_bf = (unsigned short*)alloc((size_t)NH * 128 * 2);

  auto convW = [&](const float* src, unsigned short* dst, int K, int N, int Npad) {
    int total = K * Npad;
    convert_pack_w_kernel<<<(total + 255) / 256, 256, 0, stream>>>(src, dst, K, N, Npad);
  };
  auto gemm = [&](const unsigned short* Am, int lda, const unsigned short* Wm, int Npad,
                  const float* bias, float* oF, unsigned short* oB,
                  int M, int K, int Nlog, int ldo, int relu, int mode) {
    dim3 grid(Npad / 64, M / 64);
    gemm_wmma_bf16<<<grid, 128, 0, stream>>>(Am, lda, Wm, Npad, bias, oF, oB,
                                             K, Nlog, ldo, relu, mode);
  };

  // ---- convert: obs row-major; weights pair-packed + zero-padded to N%64==0
  convert_bf16_kernel<<<(N8 * 1024 + 255) / 256, 256, 0, stream>>>(obs, obs_bf, N8 * 1024);
  convW(bbw1, wb_bb1, 1024, 512, 512);
  convW(bbw2, wb_bb2, 512,  512, 512);
  convW(polw, wb_pol, 512,  16,  64);
  convW(yw,   wb_y,   512,  600, 640);
  convW(ciw,  wb_ci,  512,  128, 128);
  convW(whh,  wb_hh,  128,  512, 512);
  convW(zw1,  wb_z1,  128,  128, 128);
  convW(zw2,  wb_z2,  128,  600, 640);
  convW(qw1,  wb_q1,  128,  128, 128);
  convW(qw2,  wb_q2,  128,  601, 640);
  convW(aw1,  wb_a1,  128,  128, 128);
  convW(aw2,  wb_a2,  128,  16,  64);

  // ---- output offsets (flat, in return order) ----
  const size_t off_logits = 0;
  const size_t off_y   = (size_t)N8 * 16;
  const size_t off_z   = off_y + (size_t)N8 * 600;
  const size_t off_q   = off_z + (size_t)N8 * 16 * 600;
  const size_t off_aux = off_q + (size_t)N8 * 16 * 601;

  // ---- backbone ----
  gemm(obs_bf, 1024, wb_bb1, 512, bbb1, nullptr, h1_bf,  N8, 1024, 512, 512, 1, 0);
  gemm(h1_bf,  512,  wb_bb2, 512, bbb2, nullptr, emb_bf, N8, 512,  512, 512, 1, 0);

  // ---- embedding heads ----
  gemm(emb_bf, 512, wb_pol, 64,  polb, out + off_logits, nullptr, N8, 512, 16,  16,  0, 1);
  gemm(emb_bf, 512, wb_y,   640, yb,   out + off_y,      nullptr, N8, 512, 600, 600, 0, 1);
  gemm(emb_bf, 512, wb_ci,  128, cib,  c_f32,            nullptr, N8, 512, 128, 128, 0, 1);
  tanh_c0_kernel<<<(N8 * 128 + 255) / 256, 256, 0, stream>>>(c_f32, h0_bf, N8 * 128);

  // ---- LSTM scan over A=16 action steps ----
  for (int a = 0; a < 16; ++a) {
    const unsigned short* h_in =
        (a == 0) ? h0_bf : (h_all + (size_t)(a - 1) * N8 * 128);
    gemm(h_in, 128, wb_hh, 512, nullptr, G_f32, nullptr, N8, 128, 512, 512, 0, 1);
    lstm_pw_kernel<<<(N8 * 128 + 255) / 256, 256, 0, stream>>>(
        G_f32, wih, bih, bhh, a, c_f32, h_all + (size_t)a * N8 * 128);
  }

  // ---- head MLPs on h_all [131072,128]; mode 2 scatters to [T,B,A,D] ----
  gemm(h_all,  128, wb_z1, 128, zb1, nullptr, hid_bf, NH, 128, 128, 128, 1, 0);
  gemm(hid_bf, 128, wb_z2, 640, zb2, out + off_z,   nullptr, NH, 128, 600, 600, 0, 2);
  gemm(h_all,  128, wb_q1, 128, qb1, nullptr, hid_bf, NH, 128, 128, 128, 1, 0);
  gemm(hid_bf, 128, wb_q2, 640, qb2, out + off_q,   nullptr, NH, 128, 601, 601, 0, 2);
  gemm(h_all,  128, wb_a1, 128, ab1, nullptr, hid_bf, NH, 128, 128, 128, 1, 0);
  gemm(hid_bf, 128, wb_a2, 64,  ab2, out + off_aux, nullptr, NH, 128, 16,  16,  0, 2);
}
